// CausalSelfAttention_12360915877935
// MI455X (gfx1250) — matrix operand order repaired and checked by reading, not compiled, not run on hardware
//
#include <hip/hip_runtime.h>

// ---------------------------------------------------------------------------
// CDNA5 (gfx1250) bf16-WMMA implementation of causal GQA self-attention block:
//   qkv = x @ [Wq|Wk|Wv]  (bf16 WMMA, fp32 accum, async double-buffered B tiles)
//   rmsnorm + rope on q,k (fp32)
//   flash-style causal attention with GQA (bf16 WMMA, fp32 online softmax,
//                                          async K-tile staging)
//   out = y @ Wc           (bf16 WMMA, fp32 accum, fp32 output)
//
// Shapes: B=4, T=2048, C=2048, H=16, HKV=4, D=128, REP=4.
// Workspace layout (bytes), total ~100 MB:
//   [0,            12582912)  Wqkv^T  bf16 [3072][2048]
//   [12582912,     20971520)  Wc^T    bf16 [2048][2048]
//   [20971520,     71303168)  qkv     bf16 [8192][3072]
//   [71303168,    104857600)  y       bf16 [8192][2048]
// ---------------------------------------------------------------------------

#define TT   2048
#define CC   2048
#define HH   16
#define HKV  4
#define DD   128
#define ROWQKV 3072   // H*D + 2*HKV*D
#define NROWS  8192   // B*T

typedef __attribute__((ext_vector_type(16))) __bf16 v16bf;
typedef __attribute__((ext_vector_type(8)))  __bf16 v8bf;
typedef __attribute__((ext_vector_type(8)))  float  v8f;

static __device__ __forceinline__ __bf16 f2bf(float f) {
    unsigned u = __builtin_bit_cast(unsigned, f);
    unsigned r = (u + 0x7FFFu + ((u >> 16) & 1u)) >> 16;
    return __builtin_bit_cast(__bf16, (unsigned short)r);
}
static __device__ __forceinline__ float bf2f(__bf16 b) {
    unsigned u = ((unsigned)__builtin_bit_cast(unsigned short, b)) << 16;
    return __builtin_bit_cast(float, u);
}

union V16U { v16bf v; v8bf h[2]; };

// A-fragment style load: elements 0..7 from lo[0..7], 8..15 from hi[0..7].
static __device__ __forceinline__ v16bf ld16_gap(const __bf16* lo, const __bf16* hi) {
    V16U u;
    u.h[0] = *(const v8bf*)lo;
    u.h[1] = *(const v8bf*)hi;
    return u.v;
}
static __device__ __forceinline__ v16bf ld16(const __bf16* p) { return ld16_gap(p, p + 8); }

static __device__ __forceinline__ v8f wmma_bf16(v16bf a, v16bf b, v8f c) {
    return __builtin_amdgcn_wmma_f32_16x16x32_bf16(
        /*neg_a=*/false, a, /*neg_b=*/false, b,
        /*c_mod=*/(short)0, c, /*reuse_a=*/false, /*reuse_b=*/false);
}

// Async 16B global->LDS copy (ASYNCcnt-tracked, no VGPR data path).
// dst: LDS byte offset (low 32 bits of flat shared pointer, per ISA LDS
// aperture truncation); src: 64-bit global address.
static __device__ __forceinline__ void async_cp16(const __bf16* gsrc, __bf16* lds_dst) {
    unsigned laddr = (unsigned)(unsigned long long)lds_dst;
    asm volatile("global_load_async_to_lds_b128 %0, %1, off"
                 :: "v"(laddr), "v"(gsrc) : "memory");
}
static __device__ __forceinline__ void wait_async0() {
#if __has_builtin(__builtin_amdgcn_s_wait_asynccnt)
    __builtin_amdgcn_s_wait_asynccnt(0);
#else
    asm volatile("s_wait_asynccnt 0x0" ::: "memory");
#endif
}

// ---------------------------------------------------------------------------
// Weight convert + transpose: out[n][k] = (bf16) in[k][n]; in is [K][N] fp32.
// ---------------------------------------------------------------------------
__global__ __launch_bounds__(256) void conv_t(const float* __restrict__ in,
                                              __bf16* __restrict__ out,
                                              int K, int N) {
    long idx = (long)blockIdx.x * 256 + threadIdx.x;
    int n = (int)(idx / K);
    int k = (int)(idx % K);
    out[idx] = f2bf(in[(size_t)k * N + n]);
}

// ---------------------------------------------------------------------------
// Tiled GEMM: C[M][N] = A[M][K] * Bt[N][K]^T.  BM=BN=128, BK=32.
// 256 threads = 8 waves; wave (mw in 0..3, nw in 0..1) owns a 32x64 tile.
// B tiles are async-DMA'd global->LDS and double-buffered so weight traffic
// overlaps the WMMA stream; A tiles are staged synchronously (f32->bf16 VALU
// conversion for the x @ Wqkv case).
// ---------------------------------------------------------------------------
template <int A_F32, int OUT_F32>
__global__ __launch_bounds__(256) void gemm_wmma(const void* __restrict__ Ap,
                                                 const __bf16* __restrict__ Bt,
                                                 void* __restrict__ Cp,
                                                 int M, int N, int K) {
    __shared__ __bf16 ldsA[128 * 40];      // pitch 40 bf16 = 80B (16B-aligned rows)
    __shared__ __bf16 ldsB[2][128 * 40];   // double-buffered async target

    const int tid  = threadIdx.x;
    const int w    = tid >> 5;
    const int lane = tid & 31;
    const int ln   = lane & 15;
    const int h16  = lane >> 4;
    const int mw   = w & 3;
    const int nw   = w >> 2;
    const int m0   = blockIdx.y * 128;
    const int n0   = blockIdx.x * 128;

    // Issue one B tile (128n x 32k bf16 = 512 x 16B chunks) asynchronously.
    auto issueB = [&](int buf, int ktile) {
#pragma unroll
        for (int it = 0; it < 2; ++it) {
            int c   = it * 256 + tid;
            int row = c >> 2, seg = c & 3;
            const __bf16* g = Bt + (size_t)(n0 + row) * K + ktile + seg * 8;
            async_cp16(g, &ldsB[buf][row * 40 + seg * 8]);
        }
    };

    v8f zero = {0.f, 0.f, 0.f, 0.f, 0.f, 0.f, 0.f, 0.f};
    v8f acc[2][4];
#pragma unroll
    for (int mt = 0; mt < 2; ++mt)
#pragma unroll
        for (int nt = 0; nt < 4; ++nt) acc[mt][nt] = zero;

    issueB(0, 0);                          // prologue: tile 0 in flight
    int cur = 0;
    for (int kt = 0; kt < K; kt += 32, cur ^= 1) {
        // Stage A tile (128x32) into LDS as bf16 (sync path).
#pragma unroll
        for (int it = 0; it < 16; ++it) {
            int flat = it * 256 + tid;
            int r = flat >> 5, c = flat & 31;
            if constexpr (A_F32) {
                float fv = ((const float*)Ap)[(size_t)(m0 + r) * K + kt + c];
                ldsA[r * 40 + c] = f2bf(fv);
            } else {
                ldsA[r * 40 + c] = ((const __bf16*)Ap)[(size_t)(m0 + r) * K + kt + c];
            }
        }
        wait_async0();                     // B[cur] has landed in LDS
        __syncthreads();
        if (kt + 32 < K) issueB(cur ^ 1, kt + 32);   // prefetch next B tile

        v16bf af[2];
#pragma unroll
        for (int mt = 0; mt < 2; ++mt) {
            const __bf16* pa = &ldsA[(mw * 32 + mt * 16 + ln) * 40];
            af[mt] = ld16_gap(pa + 8 * h16, pa + 16 + 8 * h16);
        }
        v16bf bf[4];
#pragma unroll
        for (int nt = 0; nt < 4; ++nt) {
            bf[nt] = ld16_gap(&ldsB[cur][(nw * 64 + nt * 16 + ln) * 40 + 8 * h16], &ldsB[cur][(nw * 64 + nt * 16 + ln) * 40 + 16 + 8 * h16]);
        }
#pragma unroll
        for (int mt = 0; mt < 2; ++mt)
#pragma unroll
            for (int nt = 0; nt < 4; ++nt)
                acc[mt][nt] = wmma_bf16(af[mt], bf[nt], acc[mt][nt]);
        __syncthreads();                   // protect ldsA before next staging
    }

#pragma unroll
    for (int mt = 0; mt < 2; ++mt)
#pragma unroll
        for (int nt = 0; nt < 4; ++nt)
#pragma unroll
            for (int v = 0; v < 8; ++v) {
                int row = m0 + mw * 32 + mt * 16 + v + 8 * h16;
                int col = n0 + nw * 64 + nt * 16 + ln;
                if constexpr (OUT_F32)
                    ((float*)Cp)[(size_t)row * N + col] = acc[mt][nt][v];
                else
                    ((__bf16*)Cp)[(size_t)row * N + col] = f2bf(acc[mt][nt][v]);
            }
}

// ---------------------------------------------------------------------------
// RMSNorm + RoPE on q (16 heads) and k (4 heads), in place, fp32 math.
// One wave per (token, head-slot); 4 waves per 128-thread block.
// ---------------------------------------------------------------------------
__global__ __launch_bounds__(128) void rms_rope(__bf16* __restrict__ qkv,
                                                const float* __restrict__ freqs) {
    const int w    = threadIdx.x >> 5;
    const int lane = threadIdx.x & 31;
    const long slot = (long)blockIdx.x * 4 + w;       // < NROWS * 20
    const int bt = (int)(slot / 20);
    const int hh = (int)(slot % 20);
    const int t  = bt & (TT - 1);
    __bf16* row = qkv + (size_t)bt * ROWQKV +
                  (hh < HH ? hh * DD : HH * DD + (hh - HH) * DD);

    const int c0 = lane * 4;
    float x0 = bf2f(row[c0 + 0]);
    float x1 = bf2f(row[c0 + 1]);
    float x2 = bf2f(row[c0 + 2]);
    float x3 = bf2f(row[c0 + 3]);

    float ss = x0 * x0 + x1 * x1 + x2 * x2 + x3 * x3;
#pragma unroll
    for (int off = 16; off >= 1; off >>= 1) ss += __shfl_xor(ss, off, 32);
    const float scale = rsqrtf(ss * (1.0f / 128.0f) + 1e-6f);

    const int p0 = lane * 2;                          // rope pair index
    const float* f0 = freqs + ((size_t)t * (DD / 2) + p0) * 2;
    const float c_0 = f0[0], s_0 = f0[1], c_1 = f0[2], s_1 = f0[3];
    x0 *= scale; x1 *= scale; x2 *= scale; x3 *= scale;
    row[c0 + 0] = f2bf(x0 * c_0 - x1 * s_0);
    row[c0 + 1] = f2bf(x0 * s_0 + x1 * c_0);
    row[c0 + 2] = f2bf(x2 * c_1 - x3 * s_1);
    row[c0 + 3] = f2bf(x2 * s_1 + x3 * c_1);
}

// ---------------------------------------------------------------------------
// Flash-style causal GQA attention.
// grid = (T/64, B*H); block = 128 threads = 4 waves; wave owns 16 q-rows.
// Key tiles of 32 staged into LDS: K async-DMA'd row-major, V transposed via
// the sync path (needs element scatter).
// ---------------------------------------------------------------------------
__global__ __launch_bounds__(128) void attn(const __bf16* __restrict__ qkv,
                                            __bf16* __restrict__ yb) {
    __shared__ __bf16 ldsK[32 * 136];        // [key][ch], pitch 136
    __shared__ __bf16 ldsVt[128 * 40];       // [ch][key], pitch 40
    __shared__ __bf16 ldsP[4 * 16 * 40];     // per-wave P bounce [q][key]

    const int tid  = threadIdx.x;
    const int w    = tid >> 5;
    const int lane = tid & 31;
    const int ln   = lane & 15;
    const int h16  = lane >> 4;

    const int b  = blockIdx.y >> 4;
    const int h  = blockIdx.y & 15;
    const int hk = h >> 2;                   // REP = 4
    const int qb0 = blockIdx.x * 64;
    const int qw0 = qb0 + w * 16;
    const float SCALE = 0.08838834764831843f;   // 1/sqrt(128)
    const float NEG_INF = -__builtin_inff();

    // Preload Q fragments for this wave's 16 rows (4 chunks of K=32).
    const __bf16* qrow = qkv + ((size_t)(b * TT + qw0 + ln)) * ROWQKV + h * DD;
    v16bf aq[4];
#pragma unroll
    for (int kk = 0; kk < 4; ++kk)
        aq[kk] = ld16_gap(qrow + kk * 32 + 8 * h16, qrow + kk * 32 + 16 + 8 * h16);

    v8f zero = {0.f, 0.f, 0.f, 0.f, 0.f, 0.f, 0.f, 0.f};
    v8f o[8];
    float m_i[8], l_i[8];
#pragma unroll
    for (int nc = 0; nc < 8; ++nc) o[nc] = zero;
#pragma unroll
    for (int v = 0; v < 8; ++v) { m_i[v] = NEG_INF; l_i[v] = 0.f; }

    const int nkt = (qb0 >> 5) + 2;          // key tiles needed for causal block
    for (int kt = 0; kt < nkt; ++kt) {
        const int kb0 = kt * 32;
        __syncthreads();
        // Async K tile: 32 keys x 128 ch = 512 x 16B chunks, straight copy.
#pragma unroll
        for (int it = 0; it < 4; ++it) {
            int flat = it * 128 + tid;
            int key = flat >> 4;
            int ch  = (flat & 15) * 8;
            const __bf16* src = qkv + ((size_t)(b * TT + kb0 + key)) * ROWQKV + hk * DD;
            async_cp16(src + HH * DD + ch, &ldsK[key * 136 + ch]);
        }
        // Sync V tile with transpose scatter into ldsVt.
#pragma unroll
        for (int it = 0; it < 4; ++it) {
            int flat = it * 128 + tid;
            int key = flat >> 4;
            int ch  = (flat & 15) * 8;
            const __bf16* src = qkv + ((size_t)(b * TT + kb0 + key)) * ROWQKV + hk * DD;
            v8bf vvec = *(const v8bf*)(src + HH * DD + HKV * DD + ch);
#pragma unroll
            for (int j = 0; j < 8; ++j) ldsVt[(ch + j) * 40 + key] = vvec[j];
        }
        wait_async0();
        __syncthreads();

        // S = Q @ K^T  (two 16-key column tiles).
        v8f s0 = zero, s1 = zero;
#pragma unroll
        for (int kk = 0; kk < 4; ++kk) {
            v16bf b0 = ld16_gap(&ldsK[(ln)      * 136 + kk * 32 + 8 * h16], &ldsK[(ln)      * 136 + kk * 32 + 16 + 8 * h16]);
            v16bf b1 = ld16_gap(&ldsK[(16 + ln) * 136 + kk * 32 + 8 * h16], &ldsK[(16 + ln) * 136 + kk * 32 + 16 + 8 * h16]);
            s0 = wmma_bf16(aq[kk], b0, s0);
            s1 = wmma_bf16(aq[kk], b1, s1);
        }

        // Online softmax update (fp32).
        float alpha[8];
#pragma unroll
        for (int v = 0; v < 8; ++v) {
            const int q = qw0 + v + 8 * h16;
            float a = s0[v] * SCALE;
            float c = s1[v] * SCALE;
            if (kb0 + ln > q)      a = NEG_INF;
            if (kb0 + 16 + ln > q) c = NEG_INF;
            float rm = fmaxf(a, c);
#pragma unroll
            for (int off = 8; off >= 1; off >>= 1)
                rm = fmaxf(rm, __shfl_xor(rm, off, 32));
            const float mn = fmaxf(m_i[v], rm);
            alpha[v] = __expf(m_i[v] - mn);
            const float pa = __expf(a - mn);
            const float pc = __expf(c - mn);
            float rs = pa + pc;
#pragma unroll
            for (int off = 8; off >= 1; off >>= 1)
                rs += __shfl_xor(rs, off, 32);
            l_i[v] = alpha[v] * l_i[v] + rs;
            m_i[v] = mn;
            s0[v] = pa;
            s1[v] = pc;
        }
#pragma unroll
        for (int nc = 0; nc < 8; ++nc)
#pragma unroll
            for (int v = 0; v < 8; ++v) o[nc][v] *= alpha[v];

        // P (C-layout) -> per-wave LDS -> A-layout fragment.
        __bf16* pw = &ldsP[w * 16 * 40];
#pragma unroll
        for (int v = 0; v < 8; ++v) {
            const int r = v + 8 * h16;
            pw[r * 40 + ln]      = f2bf(s0[v]);
            pw[r * 40 + 16 + ln] = f2bf(s1[v]);
        }
        asm volatile("s_wait_dscnt 0" ::: "memory");   // wave-local LDS RAW
        v16bf pf = ld16_gap(pw + ln * 40 + 8 * h16, pw + ln * 40 + 16 + 8 * h16);

        // O += P @ V  (8 channel tiles of 16).
#pragma unroll
        for (int nc = 0; nc < 8; ++nc) {
            v16bf bv = ld16_gap(&ldsVt[(nc * 16 + ln) * 40 + 8 * h16], &ldsVt[(nc * 16 + ln) * 40 + 16 + 8 * h16]);
            o[nc] = wmma_bf16(pf, bv, o[nc]);
        }
    }

    // Epilogue: normalize and store y (B,T,H,D) as bf16.
#pragma unroll
    for (int v = 0; v < 8; ++v) {
        const float inv = 1.0f / l_i[v];
        const int row = qw0 + v + 8 * h16;
        __bf16* dst = yb + ((size_t)(b * TT + row)) * (HH * DD) + h * DD;
#pragma unroll
        for (int nc = 0; nc < 8; ++nc)
            dst[nc * 16 + ln] = f2bf(o[nc][v] * inv);
    }
}

// ---------------------------------------------------------------------------
// Host launch.
// ---------------------------------------------------------------------------
extern "C" void kernel_launch(void* const* d_in, const int* in_sizes, int n_in,
                              void* d_out, int out_size, void* d_ws, size_t ws_size,
                              hipStream_t stream) {
    const float* x     = (const float*)d_in[0];
    const float* freqs = (const float*)d_in[1];
    const float* Wq    = (const float*)d_in[2];
    const float* Wk    = (const float*)d_in[3];
    const float* Wv    = (const float*)d_in[4];
    const float* Wc    = (const float*)d_in[5];
    float* out = (float*)d_out;

    char* ws = (char*)d_ws;
    __bf16* Wt_qkv = (__bf16*)(ws);                    // [3072][2048]
    __bf16* Wt_c   = (__bf16*)(ws + 12582912);         // [2048][2048]
    __bf16* qkvb   = (__bf16*)(ws + 20971520);         // [8192][3072]
    __bf16* yb     = (__bf16*)(ws + 71303168);         // [8192][2048]

    // 1) Convert + transpose weights to bf16 B-operands.
    conv_t<<<(2048 * 2048) / 256, 256, 0, stream>>>(Wq, Wt_qkv, 2048, 2048);
    conv_t<<<(512  * 2048) / 256, 256, 0, stream>>>(Wk, Wt_qkv + (size_t)2048 * 2048, 2048, 512);
    conv_t<<<(512  * 2048) / 256, 256, 0, stream>>>(Wv, Wt_qkv + (size_t)2560 * 2048, 2048, 512);
    conv_t<<<(2048 * 2048) / 256, 256, 0, stream>>>(Wc, Wt_c, 2048, 2048);

    // 2) qkv = x @ [Wq|Wk|Wv]   (M=8192, N=3072, K=2048)
    dim3 g1(3072 / 128, 8192 / 128);
    gemm_wmma<1, 0><<<g1, 256, 0, stream>>>(x, Wt_qkv, qkvb, NROWS, ROWQKV, CC);

    // 3) RMSNorm + RoPE on q,k in place.
    rms_rope<<<(NROWS * 20) / 4, 128, 0, stream>>>(qkvb, freqs);

    // 4) Causal GQA flash attention -> y.
    dim3 ga(TT / 64, 4 * HH);
    attn<<<ga, 128, 0, stream>>>(qkvb, yb);

    // 5) out = y @ Wc           (M=8192, N=2048, K=2048), fp32 output.
    dim3 g2(2048 / 128, 8192 / 128);
    gemm_wmma<0, 1><<<g2, 256, 0, stream>>>(yb, Wt_c, out, NROWS, CC, CC);
}
